// NeuralCRF_34557306864128
// MI455X (gfx1250) — compile-verified, matrix operand order checked by reading, not run on hardware
//
#include <hip/hip_runtime.h>

typedef __attribute__((ext_vector_type(16))) _Float16 v16h;
typedef __attribute__((ext_vector_type(8)))  _Float16 v8h;
typedef __attribute__((ext_vector_type(8)))  float    v8f;
typedef __attribute__((ext_vector_type(4)))  float    v4f;

#define S_LEN   1024
#define T_TAGS  128
#define B_BATCH 64
#define ROWS    16      // batches per workgroup = WMMA M
#define BSTRIDE 132     // Beta row stride (floats): 528B = 33*16, bank-spread
#define WSTRIDE 136     // w row stride (halves): 272B = 17*16, bank-spread

#define LOG2E 1.4426950408889634f
#define LN2   0.6931471805599453f

// Backward CRF DP in the log2 domain (Beta2 = log2e * Beta):
//   v2      = em*log2e + Beta2            (one v_fma per element)
//   M2      = rowmax(v2)
//   w       = exp2(v2 - M2)               (raw v_exp_f32)
//   s[b,i]  = sum_j w[b,j] * exp(T[i,j])  (WMMA f16xf16->f32, K=128)
//   Beta2'  = M2 + log2(s)                (raw v_log_f32)
// Identical math to lse over (T + em + Beta): 2^(v2-M2) = e^(v-M), and
// M2 + log2(s) = log2e*(M + ln s).  s in [0.9, 142] -> raw trans ops safe.
__global__ __launch_bounds__(256, 1)
void crf_normalizer_kernel(const int*   __restrict__ Wtok,
                           const float* __restrict__ em,
                           const float* __restrict__ trans,
                           float*       __restrict__ out)
{
    __shared__ __align__(16) float    Beta[ROWS][BSTRIDE];   // log2 domain
    __shared__ __align__(16) _Float16 wbuf[ROWS][WSTRIDE];
    __shared__ unsigned maskbits[S_LEN];
    __shared__ __align__(16) float Mrow[ROWS];

    const int tid    = threadIdx.x;
    const int wave   = tid >> 5;          // 0..7 : output tag tile (N)
    const int lane   = tid & 31;
    const int half   = lane >> 4;         // 0/1 : lane half (K sub-range)
    const int lane16 = lane & 15;
    const int b0     = blockIdx.x * ROWS;

    // ---------------- one-time init ----------------
    for (int i = tid; i < ROWS * BSTRIDE; i += 256)
        ((float*)Beta)[i] = 0.0f;                       // Beta0 = 0
    for (int t = tid; t < S_LEN; t += 256) {            // packed mask bits
        unsigned wb = 0;
        for (int m = 0; m < ROWS; ++m) {
            int tok = Wtok[(size_t)(b0 + m) * S_LEN + t];
            if (tok != 0 && tok != 3) wb |= (1u << m);  // !PAD && !EOS
        }
        maskbits[t] = wb;
    }

    // Resident B fragments: wave nt owns tags i in [nt*16, nt*16+16).
    // 16-bit fragment K packing (CDNA5 ISA): lane<16 -> K{0..7,16..23},
    // lane>=16 -> K{8..15,24..31}; per-column layout assumed symmetric for B.
    v16h bfrag[4];
    {
        const float* trow = trans + (size_t)(wave * 16 + lane16) * T_TAGS;
        #pragma unroll
        for (int kc = 0; kc < 4; ++kc) {
            const int base = kc * 32 + half * 8;
            #pragma unroll
            for (int e = 0; e < 8; ++e) {
                bfrag[kc][e]     = (_Float16)__builtin_amdgcn_exp2f(trow[base + e]      * LOG2E);
                bfrag[kc][e + 8] = (_Float16)__builtin_amdgcn_exp2f(trow[base + 16 + e] * LOG2E);
            }
        }
    }
    __syncthreads();

    // ---------------- backward DP: t = S-1 .. 1 ----------------
    const int mrow1 = 2 * wave + half;    // row this (wave,half) owns in phase 1
    const int j0    = lane16 * 8;         // 8 tag columns per lane

    #pragma unroll 1
    for (int t = S_LEN - 1; t >= 1; --t) {
        // phase 1: v2 = em*log2e + Beta2 ; M2 = rowmax ; w = exp2(v2-M2) -> f16
        const float* ep = em + ((size_t)(b0 + mrow1) * S_LEN + t) * T_TAGS + j0;
        v4f e0 = *(const v4f*)(ep);
        v4f e1 = *(const v4f*)(ep + 4);
        v4f bv0 = *(const v4f*)&Beta[mrow1][j0];
        v4f bv1 = *(const v4f*)&Beta[mrow1][j0 + 4];
        float v[8];
        #pragma unroll
        for (int k = 0; k < 4; ++k) {
            v[k]     = __builtin_fmaf(e0[k], LOG2E, bv0[k]);
            v[k + 4] = __builtin_fmaf(e1[k], LOG2E, bv1[k]);
        }
        float mx = v[0];
        #pragma unroll
        for (int k = 1; k < 8; ++k) mx = fmaxf(mx, v[k]);
        #pragma unroll
        for (int off = 1; off < 16; off <<= 1)
            mx = fmaxf(mx, __shfl_xor(mx, off, 16));    // 16-lane row max
        v8h wv;
        #pragma unroll
        for (int k = 0; k < 8; ++k)
            wv[k] = (_Float16)__builtin_amdgcn_exp2f(v[k] - mx);
        *(v8h*)&wbuf[mrow1][j0] = wv;                   // ds_store_b128
        if (lane16 == 0) Mrow[mrow1] = mx;
        if (t > 1) __builtin_prefetch(ep - T_TAGS, 0, 1); // next emission row

        __syncthreads();

        // phase 2: s = w x E^T via 4 chained WMMAs (K=128); branchless update
        v8f acc = {};
        #pragma unroll
        for (int kc = 0; kc < 4; ++kc) {
            const int base = kc * 32 + half * 8;
            v8h lo = *(const v8h*)&wbuf[lane16][base];        // A row = lane16
            v8h hi = *(const v8h*)&wbuf[lane16][base + 16];
            v16h a = __builtin_shufflevector(lo, hi,
                        0,1,2,3,4,5,6,7,8,9,10,11,12,13,14,15);
            acc = __builtin_amdgcn_wmma_f32_16x16x32_f16(
                      false, a, false, bfrag[kc], (short)0, acc, false, false);
        }
        const unsigned mbits = maskbits[t] >> (half * 8);
        const int col = wave * 16 + lane16;
        v4f m0 = *(const v4f*)&Mrow[half * 8];
        v4f m1 = *(const v4f*)&Mrow[half * 8 + 4];
        float mv[8];
        #pragma unroll
        for (int k = 0; k < 4; ++k) { mv[k] = m0[k]; mv[k + 4] = m1[k]; }
        #pragma unroll
        for (int r = 0; r < 8; ++r) {                   // D: VGPR r -> row r+8*half
            const int mr = r + half * 8;
            const float oldv = Beta[mr][col];
            const float newv = mv[r] + __builtin_amdgcn_logf(acc[r]); // raw log2
            Beta[mr][col] = ((mbits >> r) & 1u) ? newv : oldv;        // cndmask
        }
        __syncthreads();
    }

    // ---- finalize: logZ = lse_i( trans[BOT=1,i] + em[:,0,i] + ln2*Beta2 ) ----
    {
        const float* ep = em + (size_t)(b0 + mrow1) * S_LEN * T_TAGS + j0;
        const float* tp = trans + 1 * T_TAGS + j0;      // BOT_IDX = 1
        float x[8];
        #pragma unroll
        for (int k = 0; k < 8; ++k)
            x[k] = tp[k] + ep[k] + Beta[mrow1][j0 + k] * LN2;
        float mx = x[0];
        #pragma unroll
        for (int k = 1; k < 8; ++k) mx = fmaxf(mx, x[k]);
        #pragma unroll
        for (int off = 1; off < 16; off <<= 1)
            mx = fmaxf(mx, __shfl_xor(mx, off, 16));
        float s = 0.0f;
        #pragma unroll
        for (int k = 0; k < 8; ++k)
            s += __builtin_amdgcn_exp2f((x[k] - mx) * LOG2E);
        #pragma unroll
        for (int off = 1; off < 16; off <<= 1)
            s += __shfl_xor(s, off, 16);
        if (lane16 == 0)
            out[b0 + mrow1] = mx + LN2 * __builtin_amdgcn_logf(s);
    }
}

extern "C" void kernel_launch(void* const* d_in, const int* in_sizes, int n_in,
                              void* d_out, int out_size, void* d_ws, size_t ws_size,
                              hipStream_t stream) {
    const int*   Wtok  = (const int*)d_in[0];     // [64,1024] token ids
    const float* em    = (const float*)d_in[1];   // [64,1024,128] f32
    const float* trans = (const float*)d_in[2];   // [128,128] f32
    float*       out   = (float*)d_out;           // [64] f32 logZ
    (void)in_sizes; (void)n_in; (void)out_size; (void)d_ws; (void)ws_size;
    crf_normalizer_kernel<<<B_BATCH / ROWS, 256, 0, stream>>>(Wtok, em, trans, out);
}
